// EquivariantTransformerBlock_29214367547539
// MI455X (gfx1250) — matrix operand order repaired and checked by reading, not compiled, not run on hardware
//
#include <hip/hip_runtime.h>
#include <math.h>
#include <stdint.h>

// ---------------- problem constants ----------------
#define NN      25000
#define EE      400000
#define HID     128
#define EDGE_F  16
#define HEADS   8
#define ADY     64
#define INNER   160
#define MSG_F   273
#define KPAD1   288           // 273 padded to 9*32
#define KT1     9
#define NT1     10
#define KT2     5
#define NT2     10
#define KTV     5
#define NTV     8
#define TILES   (EE/16)

typedef __attribute__((ext_vector_type(16))) _Float16 v16h;
typedef __attribute__((ext_vector_type(8)))  _Float16 v8h;
typedef __attribute__((ext_vector_type(8)))  float    v8f;

// ---------------- workspace layout -----------------
static constexpr size_t alup(size_t x) { return (x + 255) & ~(size_t)255; }
static constexpr size_t SZ_M2SW = (size_t)TILES * KT2 * 512 * 2;
static constexpr size_t SZ_WE1  = (size_t)KT1 * NT1 * 512 * 2;
static constexpr size_t SZ_WE2  = (size_t)KT2 * NT2 * 512 * 2;
static constexpr size_t SZ_WV   = (size_t)KTV * NTV * 512 * 2;
static constexpr size_t SZ_LOG  = (size_t)EE * HEADS * 4;
static constexpr size_t SZ_EX   = SZ_LOG;
static constexpr size_t SZ_MX   = (size_t)NN * HEADS * 4;
static constexpr size_t SZ_DEN  = SZ_MX;
static constexpr size_t SZ_CAC  = (size_t)NN * 3 * 4;
static constexpr size_t SZ_HAG  = (size_t)NN * HID * 4;
static constexpr size_t SZ_H16  = (size_t)NN * HID * 2;
static constexpr size_t SZ_A16  = (size_t)EE * EDGE_F * 2;

static constexpr size_t O_M2  = 0;
static constexpr size_t O_WE1 = O_M2  + alup(SZ_M2SW);
static constexpr size_t O_WE2 = O_WE1 + alup(SZ_WE1);
static constexpr size_t O_WV  = O_WE2 + alup(SZ_WE2);
static constexpr size_t O_LOG = O_WV  + alup(SZ_WV);
static constexpr size_t O_EX  = O_LOG + alup(SZ_LOG);
static constexpr size_t O_MX  = O_EX  + alup(SZ_EX);
static constexpr size_t O_DEN = O_MX  + alup(SZ_MX);
static constexpr size_t O_CAC = O_DEN + alup(SZ_DEN);
static constexpr size_t O_HAG = O_CAC + alup(SZ_CAC);
static constexpr size_t O_H16 = O_HAG + alup(SZ_HAG);
static constexpr size_t O_A16 = O_H16 + alup(SZ_H16);

// ---------------- helpers --------------------------
__device__ __forceinline__ unsigned encf(float f) {
    unsigned u = __float_as_uint(f);
    return (u & 0x80000000u) ? ~u : (u | 0x80000000u);
}
__device__ __forceinline__ float decf(unsigned u) {
    return __uint_as_float((u & 0x80000000u) ? (u ^ 0x80000000u) : ~u);
}
__device__ __forceinline__ float siluf(float x) { return x / (1.0f + __expf(-x)); }

// assemble a 16-half A-fragment from a row: two contiguous 8-half runs
__device__ __forceinline__ v16h afrag(const _Float16* row, int kt, int g) {
    v8h lo = *(const v8h*)(row + kt * 32 + g * 16);
    v8h hi = *(const v8h*)(row + kt * 32 + g * 16 + 8);
    return __builtin_shufflevector(lo, hi, 0,1,2,3,4,5,6,7,8,9,10,11,12,13,14,15);
}

// ============================================================
// Kernel 0: fp32 -> f16 bulk convert (h, a)
// ============================================================
__global__ void k_tohalf(const float* __restrict__ s, _Float16* __restrict__ d, long n) {
    long stride = (long)gridDim.x * blockDim.x;
    for (long i = (long)blockIdx.x * blockDim.x + threadIdx.x; i < n; i += stride)
        d[i] = (_Float16)s[i];
}

// ============================================================
// Kernel 1: weights -> f16, WMMA B-fragment order
//   lane l, elem i of (kt,nt): K = kt*32 + (l>>4)*16 + i, Nc = nt*16 + (l&15)
//   remap!=0 (We1 only): permute K rows so A-tile cols are
//   [0..255]=h_src,h_dst  [256..271]=a  [272]=radial  (16B-aligned async dsts)
// ============================================================
__global__ void k_swizzle(const float* __restrict__ W, _Float16* __restrict__ dst,
                          int Krows, int Ncols, int KT, int NT, int remap) {
    int total  = KT * NT * 512;
    int stride = gridDim.x * blockDim.x;
    for (int idx = blockIdx.x * blockDim.x + threadIdx.x; idx < total; idx += stride) {
        int i    = idx & 15;
        int lane = (idx >> 4) & 31;
        int nt   = (idx >> 9) % NT;
        int kt   = idx / (512 * NT);
        int k    = kt * 32 + (lane >> 4) * 16 + i;
        if (remap) {
            if (k >= 256 && k <= 271)      k = k + 1;   // a features (src rows 257..272)
            else if (k == 272)             k = 256;     // radial     (src row 256)
        }
        float v  = (k < Krows) ? W[(size_t)k * Ncols + (nt * 16 + (lane & 15))] : 0.0f;
        dst[idx] = (_Float16)v;
    }
}

// ============================================================
// Kernel 2: init accumulators
// ============================================================
__global__ void k_init(float* __restrict__ hagg, float* __restrict__ den,
                       unsigned* __restrict__ mx, float* __restrict__ cacc) {
    int stride = gridDim.x * blockDim.x;
    for (int i = blockIdx.x * blockDim.x + threadIdx.x; i < NN * HID; i += stride) {
        hagg[i] = 0.0f;
        if (i < NN * HEADS) { den[i] = 0.0f; mx[i] = 0x007FFFFFu; }
        if (i < NN * 3) cacc[i] = 0.0f;
    }
}

// ============================================================
// Kernel 3: edge MLP (WMMA) with async global->LDS gather
// One wave32 per 16-edge tile; 4 waves / 128-thread block.
// ============================================================
__launch_bounds__(128)
__global__ void k_edge(const _Float16* __restrict__ h16, const _Float16* __restrict__ a16,
                       const float* __restrict__ coords,
                       const int* __restrict__ src, const int* __restrict__ dst,
                       const _Float16* __restrict__ we1sw, const float* __restrict__ be1,
                       const _Float16* __restrict__ we2sw, const float* __restrict__ be2,
                       const float* __restrict__ Wc, const float* __restrict__ Wa,
                       _Float16* __restrict__ m2sw, float* __restrict__ logits,
                       unsigned* __restrict__ mx, float* __restrict__ cacc) {
    __shared__ __align__(16) _Float16 A_s[4][16][KPAD1];   // gather tile / later m2
    __shared__ __align__(16) _Float16 M1_s[4][16][INNER];

    const int lane = threadIdx.x & 31;
    const int w    = threadIdx.x >> 5;
    const int tile = blockIdx.x * 4 + w;
    if (tile >= TILES) return;

    const int m = lane & 15;
    const int g = lane >> 4;
    const int e = tile * 16 + m;
    const int si = src[e], di = dst[e];

    float dx = coords[si * 3 + 0] - coords[di * 3 + 0];
    float dy = coords[si * 3 + 1] - coords[di * 3 + 1];
    float dz = coords[si * 3 + 2] - coords[di * 3 + 2];
    float radial = dx * dx + dy * dy + dz * dz;

    // ---- async gather: raw f16 rows -> LDS A tile (ASYNCcnt path) ----
    const unsigned ldsBase = (unsigned)(uintptr_t)&A_s[w][0][0];
    const uint64_t hbase = (uint64_t)(uintptr_t)h16;
    #pragma unroll
    for (int c = 0; c < 8; ++c) {                    // h[src] : 16 rows x 256B
        int ci = c * 32 + lane;
        int row = ci >> 4, seg = ci & 15;
        int srow = __shfl(si, row, 32);
        uint64_t ga = hbase + (uint64_t)srow * (HID * 2) + (unsigned)(seg * 16);
        unsigned la = ldsBase + (unsigned)(row * (KPAD1 * 2) + seg * 16);
        asm volatile("global_load_async_to_lds_b128 %0, %1, off" :: "v"(la), "v"(ga) : "memory");
    }
    #pragma unroll
    for (int c = 0; c < 8; ++c) {                    // h[dst] : 16 rows x 256B
        int ci = c * 32 + lane;
        int row = ci >> 4, seg = ci & 15;
        int drow = __shfl(di, row, 32);
        uint64_t ga = hbase + (uint64_t)drow * (HID * 2) + (unsigned)(seg * 16);
        unsigned la = ldsBase + (unsigned)(row * (KPAD1 * 2) + 256 + seg * 16);
        asm volatile("global_load_async_to_lds_b128 %0, %1, off" :: "v"(la), "v"(ga) : "memory");
    }
    {                                                // a : 16 rows x 32B -> cols 256..271
        int row = lane >> 1, part = lane & 1;
        uint64_t ga = (uint64_t)(uintptr_t)a16 + (uint64_t)(tile * 16 + row) * (EDGE_F * 2)
                      + (unsigned)(part * 16);
        unsigned la = ldsBase + (unsigned)(row * (KPAD1 * 2) + 512 + part * 16);
        asm volatile("global_load_async_to_lds_b128 %0, %1, off" :: "v"(la), "v"(ga) : "memory");
    }
    if (g == 1) {                                    // radial @272, zero pad 273..287
        A_s[w][m][272] = (_Float16)radial;
        for (int k = 273; k < KPAD1; ++k) A_s[w][m][k] = (_Float16)0.0f;
    }
    asm volatile("s_wait_asynccnt 0x0" ::: "memory");
    __builtin_amdgcn_wave_barrier();

    const v16h* B1 = (const v16h*)we1sw;
    const v16h* B2 = (const v16h*)we2sw;

    // ---- GEMM1: [16x288] @ We1sw -> silu -> M1 ----
    for (int nt = 0; nt < NT1; ++nt) {
        v8f acc;
        float bias = be1[nt * 16 + m];
        #pragma unroll
        for (int j = 0; j < 8; ++j) acc[j] = bias;
        for (int kt = 0; kt < KT1; ++kt) {
            v16h bv = B1[(kt * NT1 + nt) * 32 + lane];
            v16h av = afrag(&A_s[w][m][0], kt, g);
            acc = __builtin_amdgcn_wmma_f32_16x16x32_f16(false, av, false, bv,
                                                         (short)0, acc, false, false);
        }
        #pragma unroll
        for (int j = 0; j < 8; ++j)
            M1_s[w][j + 8 * g][nt * 16 + m] = (_Float16)siluf(acc[j]);
    }
    __builtin_amdgcn_wave_barrier();

    // ---- GEMM2: [16x160] @ We2sw -> silu -> m2 (reuse A_s cols 0..159) ----
    for (int nt = 0; nt < NT2; ++nt) {
        v8f acc;
        float bias = be2[nt * 16 + m];
        #pragma unroll
        for (int j = 0; j < 8; ++j) acc[j] = bias;
        for (int kt = 0; kt < KT2; ++kt) {
            v16h bv = B2[(kt * NT2 + nt) * 32 + lane];
            v16h av = afrag(&M1_s[w][m][0], kt, g);
            acc = __builtin_amdgcn_wmma_f32_16x16x32_f16(false, av, false, bv,
                                                         (short)0, acc, false, false);
        }
        #pragma unroll
        for (int j = 0; j < 8; ++j)
            A_s[w][j + 8 * g][nt * 16 + m] = (_Float16)siluf(acc[j]);
    }
    __builtin_amdgcn_wave_barrier();

    // ---- store m2 to global in A-fragment order (contiguous 32B/lane) ----
    v16h* M2G = (v16h*)m2sw;
    for (int kt = 0; kt < KT2; ++kt)
        M2G[((size_t)tile * KT2 + kt) * 32 + lane] = afrag(&A_s[w][m][0], kt, g);

    // ---- logits = m2 @ Wa (lane pair splits heads), coef = m2 @ Wc ----
    float lg[4] = {0.f, 0.f, 0.f, 0.f};
    float cf = 0.f;
    for (int k = 0; k < INNER; ++k) {
        float mv = (float)A_s[w][m][k];
        #pragma unroll
        for (int hh = 0; hh < 4; ++hh) lg[hh] += mv * Wa[k * HEADS + g * 4 + hh];
        if (g == 0) cf += mv * Wc[k];
    }
    #pragma unroll
    for (int hh = 0; hh < 4; ++hh) {
        int hd = g * 4 + hh;
        logits[(size_t)e * HEADS + hd] = lg[hh];
        atomicMax(&mx[(size_t)di * HEADS + hd], encf(lg[hh]));
    }
    if (g == 0) {
        float sc = cf / (sqrtf(radial + 1e-5f) + 1.0f);
        atomicAdd(&cacc[di * 3 + 0], sc * dx);
        atomicAdd(&cacc[di * 3 + 1], sc * dy);
        atomicAdd(&cacc[di * 3 + 2], sc * dz);
    }
}

// ============================================================
// Kernel 4: softmax exp + denominator
// ============================================================
__global__ void k_soft(const float* __restrict__ logits, const int* __restrict__ dst,
                       const unsigned* __restrict__ mx,
                       float* __restrict__ ex, float* __restrict__ den) {
    int stride = gridDim.x * blockDim.x;
    for (int i = blockIdx.x * blockDim.x + threadIdx.x; i < EE * HEADS; i += stride) {
        int e = i >> 3, hd = i & 7;
        int d = dst[e];
        float v = __expf(logits[i] - decf(mx[(size_t)d * HEADS + hd]));
        ex[i] = v;
        atomicAdd(&den[(size_t)d * HEADS + hd], v);
    }
}

// ============================================================
// Kernel 5: values = (m2 @ Wv) * w, scatter-add (WMMA)
// ============================================================
__launch_bounds__(256)
__global__ void k_value(const _Float16* __restrict__ m2sw, const _Float16* __restrict__ wvsw,
                        const int* __restrict__ dst, const float* __restrict__ ex,
                        const float* __restrict__ den, float* __restrict__ hagg) {
    const int lane = threadIdx.x & 31;
    const int w    = threadIdx.x >> 5;
    const int tile = blockIdx.x * 8 + w;
    if (tile >= TILES) return;
    const int g = lane >> 4, n = lane & 15;

    int dste[8];
    #pragma unroll
    for (int j = 0; j < 8; ++j) dste[j] = dst[tile * 16 + j + 8 * g];

    const v16h* AG = (const v16h*)m2sw;
    const v16h* BG = (const v16h*)wvsw;

    for (int nt = 0; nt < NTV; ++nt) {               // nt == head
        v8f acc;
        #pragma unroll
        for (int j = 0; j < 8; ++j) acc[j] = 0.0f;
        for (int kt = 0; kt < KTV; ++kt) {
            v16h av = AG[((size_t)tile * KTV + kt) * 32 + lane];
            v16h bv = BG[(kt * NTV + nt) * 32 + lane];
            acc = __builtin_amdgcn_wmma_f32_16x16x32_f16(false, av, false, bv,
                                                         (short)0, acc, false, false);
        }
        #pragma unroll
        for (int j = 0; j < 8; ++j) {
            int e = tile * 16 + j + 8 * g;
            float wgt = ex[(size_t)e * HEADS + nt] / den[(size_t)dste[j] * HEADS + nt];
            atomicAdd(&hagg[(size_t)dste[j] * HID + nt * 16 + n], acc[j] * wgt);
        }
    }
}

// ============================================================
// Kernel 6: node update (Wo GEMV, adaLN/FiLM, node MLP)
// ============================================================
__launch_bounds__(256)
__global__ void k_node(const float* __restrict__ h, const float* __restrict__ coords,
                       const float* __restrict__ y, const float* __restrict__ res,
                       const float* __restrict__ hagg, const float* __restrict__ cacc,
                       const float* __restrict__ Wo,
                       const float* __restrict__ Wn1, const float* __restrict__ bn1,
                       const float* __restrict__ Wn2, const float* __restrict__ bn2,
                       const float* __restrict__ Wf1, const float* __restrict__ bf1,
                       const float* __restrict__ Wf2, const float* __restrict__ bf2,
                       float* __restrict__ out_h2, float* __restrict__ out_coords,
                       float* __restrict__ out_res2) {
    __shared__ float hl[8][HID];
    __shared__ float tl[8][INNER];
    __shared__ float yl[8][ADY];
    const int lane = threadIdx.x & 31;
    const int w    = threadIdx.x >> 5;
    const int node = blockIdx.x * 8 + w;
    if (node >= NN) return;

    yl[w][lane]      = y[(size_t)node * ADY + lane];
    yl[w][lane + 32] = y[(size_t)node * ADY + lane + 32];
    #pragma unroll
    for (int t = 0; t < 4; ++t) hl[w][lane * 4 + t] = hagg[(size_t)node * HID + lane * 4 + t];
    __builtin_amdgcn_wave_barrier();

    float f2[4] = {0.f, 0.f, 0.f, 0.f};
    for (int k = 0; k < HID; ++k) {
        float hv = hl[w][k];
        #pragma unroll
        for (int t = 0; t < 4; ++t) f2[t] += hv * Wo[k * HID + lane * 4 + t];
    }
    float res1[4], x1[4];
    #pragma unroll
    for (int t = 0; t < 4; ++t) {
        int c = lane * 4 + t;
        res1[t] = res[(size_t)node * HID + c] + f2[t];
        x1[t]   = h[(size_t)node * HID + c] + f2[t];
    }
    float s = 0.f, s2 = 0.f;
    #pragma unroll
    for (int t = 0; t < 4; ++t) { s += x1[t]; s2 += x1[t] * x1[t]; }
    for (int off = 16; off >= 1; off >>= 1) { s += __shfl_xor(s, off, 32); s2 += __shfl_xor(s2, off, 32); }
    float mu = s / 128.f, var = s2 / 128.f - mu * mu;
    float rstd = rsqrtf(var + 1e-5f);
    float h1v[4];
    #pragma unroll
    for (int t = 0; t < 4; ++t) {
        int c = lane * 4 + t;
        float scale = bf1[c], shift = bf1[128 + c];
        for (int k = 0; k < ADY; ++k) {
            float yv = yl[w][k];
            scale += yv * Wf1[k * 256 + c];
            shift += yv * Wf1[k * 256 + 128 + c];
        }
        h1v[t] = (x1[t] - mu) * rstd * (1.f + scale) + shift;
    }
    #pragma unroll
    for (int t = 0; t < 4; ++t) hl[w][lane * 4 + t] = h1v[t];
    __builtin_amdgcn_wave_barrier();

    for (int jj = 0; jj < 5; ++jj) {
        int j = lane * 5 + jj;
        float acc = bn1[j];
        for (int k = 0; k < HID; ++k) acc += hl[w][k] * Wn1[k * INNER + j];
        tl[w][j] = siluf(acc);
    }
    __builtin_amdgcn_wave_barrier();

    float f3[4];
    #pragma unroll
    for (int t = 0; t < 4; ++t) f3[t] = bn2[lane * 4 + t];
    for (int k = 0; k < INNER; ++k) {
        float tv = tl[w][k];
        #pragma unroll
        for (int t = 0; t < 4; ++t) f3[t] += tv * Wn2[k * HID + lane * 4 + t];
    }
    float x2[4], r2v[4];
    #pragma unroll
    for (int t = 0; t < 4; ++t) { r2v[t] = res1[t] + f3[t]; x2[t] = h1v[t] + f3[t]; }
    s = 0.f; s2 = 0.f;
    #pragma unroll
    for (int t = 0; t < 4; ++t) { s += x2[t]; s2 += x2[t] * x2[t]; }
    for (int off = 16; off >= 1; off >>= 1) { s += __shfl_xor(s, off, 32); s2 += __shfl_xor(s2, off, 32); }
    float mu2 = s / 128.f, var2 = s2 / 128.f - mu2 * mu2;
    float rstd2 = rsqrtf(var2 + 1e-5f);
    #pragma unroll
    for (int t = 0; t < 4; ++t) {
        int c = lane * 4 + t;
        float scale = bf2[c], shift = bf2[128 + c];
        for (int k = 0; k < ADY; ++k) {
            float yv = yl[w][k];
            scale += yv * Wf2[k * 256 + c];
            shift += yv * Wf2[k * 256 + 128 + c];
        }
        out_h2[(size_t)node * HID + c]   = (x2[t] - mu2) * rstd2 * (1.f + scale) + shift;
        out_res2[(size_t)node * HID + c] = r2v[t];
    }
    if (lane < 3) out_coords[node * 3 + lane] = coords[node * 3 + lane] + cacc[node * 3 + lane];
}

// ============================================================
extern "C" void kernel_launch(void* const* d_in, const int* in_sizes, int n_in,
                              void* d_out, int out_size, void* d_ws, size_t ws_size,
                              hipStream_t stream) {
    const float* h      = (const float*)d_in[0];
    const float* coords = (const float*)d_in[1];
    const float* a      = (const float*)d_in[2];
    const float* y      = (const float*)d_in[3];
    const float* res    = (const float*)d_in[4];
    const int*   src    = (const int*)d_in[5];
    const int*   dst    = (const int*)d_in[6];
    const float* We1    = (const float*)d_in[7];
    const float* be1    = (const float*)d_in[8];
    const float* We2    = (const float*)d_in[9];
    const float* be2    = (const float*)d_in[10];
    const float* Wc     = (const float*)d_in[11];
    const float* Wv     = (const float*)d_in[12];
    const float* Wa     = (const float*)d_in[13];
    const float* Wo     = (const float*)d_in[14];
    const float* Wn1    = (const float*)d_in[15];
    const float* bn1    = (const float*)d_in[16];
    const float* Wn2    = (const float*)d_in[17];
    const float* bn2    = (const float*)d_in[18];
    const float* Wf1    = (const float*)d_in[19];
    const float* bf1    = (const float*)d_in[20];
    const float* Wf2    = (const float*)d_in[21];
    const float* bf2    = (const float*)d_in[22];

    char* ws = (char*)d_ws;
    _Float16* m2sw   = (_Float16*)(ws + O_M2);
    _Float16* we1sw  = (_Float16*)(ws + O_WE1);
    _Float16* we2sw  = (_Float16*)(ws + O_WE2);
    _Float16* wvsw   = (_Float16*)(ws + O_WV);
    float*    logits = (float*)(ws + O_LOG);
    float*    ex     = (float*)(ws + O_EX);
    unsigned* mx     = (unsigned*)(ws + O_MX);
    float*    den    = (float*)(ws + O_DEN);
    float*    cacc   = (float*)(ws + O_CAC);
    float*    hagg   = (float*)(ws + O_HAG);
    _Float16* h16    = (_Float16*)(ws + O_H16);
    _Float16* a16    = (_Float16*)(ws + O_A16);

    float* out_h2     = (float*)d_out;
    float* out_coords = out_h2 + (size_t)NN * HID;
    float* out_res2   = out_coords + (size_t)NN * 3;

    // 0) bulk f32->f16 (halves gather bandwidth; enables raw async copies)
    k_tohalf<<<2048, 256, 0, stream>>>(h, h16, (long)NN * HID);
    k_tohalf<<<4096, 256, 0, stream>>>(a, a16, (long)EE * EDGE_F);
    // 1) weight swizzles
    k_swizzle<<<64, 256, 0, stream>>>(We1, we1sw, MSG_F, INNER, KT1, NT1, 1);
    k_swizzle<<<64, 256, 0, stream>>>(We2, we2sw, INNER, INNER, KT2, NT2, 0);
    k_swizzle<<<64, 256, 0, stream>>>(Wv,  wvsw,  INNER, HID,   KTV, NTV, 0);
    // 2) init accumulators
    k_init<<<2048, 256, 0, stream>>>(hagg, den, mx, cacc);
    // 3) edge MLP (WMMA + async gather)
    k_edge<<<(TILES + 3) / 4, 128, 0, stream>>>(h16, a16, coords, src, dst,
                                                we1sw, be1, we2sw, be2, Wc, Wa,
                                                m2sw, logits, mx, cacc);
    // 4) softmax
    k_soft<<<4096, 256, 0, stream>>>(logits, dst, mx, ex, den);
    // 5) value aggregation (WMMA)
    k_value<<<(TILES + 7) / 8, 256, 0, stream>>>(m2sw, wvsw, dst, ex, den, hagg);
    // 6) node update
    k_node<<<(NN + 7) / 8, 256, 0, stream>>>(h, coords, y, res, hagg, cacc,
                                             Wo, Wn1, bn1, Wn2, bn2,
                                             Wf1, bf1, Wf2, bf2,
                                             out_h2, out_coords, out_res2);
}
